// RouteBlock_22746146799628
// MI455X (gfx1250) — matrix operand order repaired and checked by reading, not compiled, not run on hardware
//
#include <hip/hip_runtime.h>
#include <hip/hip_bf16.h>
#include <math.h>

// ---------------------------------------------------------------------------
// Routed MLP for MI455X (gfx1250, wave32, WMMA bf16 16x16x32, f32 accum)
// ---------------------------------------------------------------------------

#define D_MODEL 768
#define N_TOK   2048        // B*T = 2*1024
#define FCHUNK  128         // d_ff processed per chunk (8 waves * 16 cols)

// Padded LDS pitches (elements). Chosen so (pitch*2/4) mod 64 generates 16
// distinct banks across the 16 lanes of a fragment read, while every
// fragment access stays 16-byte aligned.
#define XPITCH  776         // x tile row pitch   (1552 B)
#define HPITCH  136         // h tile row pitch   ( 272 B)
#define KPITCH  40          // transposed weight column pitch (80 B)

typedef __attribute__((ext_vector_type(16))) __bf16 bf16x16;
typedef __attribute__((ext_vector_type(8)))  __bf16 bf16x8;
typedef __attribute__((ext_vector_type(8)))  float  floatx8;

// ---------------------------------------------------------------------------
// 1) Deterministic, order-preserving routing (trivial work: 2048 items)
// ---------------------------------------------------------------------------
__global__ void route_kernel(const int* __restrict__ mask,
                             int* __restrict__ big_idx,
                             int* __restrict__ small_idx,
                             int* __restrict__ counts) {
    if (blockIdx.x == 0 && threadIdx.x == 0) {
        int nb = 0, ns = 0;
        for (int t = 0; t < N_TOK; ++t) {
            if (mask[t] != 0) big_idx[nb++] = t;
            else              small_idx[ns++] = t;
        }
        counts[0] = nb;
        counts[1] = ns;
    }
}

// ---------------------------------------------------------------------------
// 2) Cast activations fp32 -> bf16 once (3 MB)
// ---------------------------------------------------------------------------
__global__ void cast_x_kernel(const float* __restrict__ x,
                              __bf16* __restrict__ xb, int n) {
    int i = blockIdx.x * blockDim.x + threadIdx.x;
    if (i < n) xb[i] = (__bf16)x[i];
}

// ---------------------------------------------------------------------------
// 3) Seed output with the proj bias selected by the mask
// ---------------------------------------------------------------------------
__global__ void init_out_kernel(float* __restrict__ out,
                                const int* __restrict__ mask,
                                const float* __restrict__ b_proj_s,
                                const float* __restrict__ b_proj_b) {
    int i = blockIdx.x * blockDim.x + threadIdx.x;
    if (i < N_TOK * D_MODEL) {
        int tok = i / D_MODEL;
        int col = i - tok * D_MODEL;
        out[i] = (mask[tok] != 0) ? b_proj_b[col] : b_proj_s[col];
    }
}

// ---------------------------------------------------------------------------
// Fragment builders following the CDNA5 16-bit WMMA VGPR layouts.
// A (16x32, MxK): lane<16 -> M=lane,   K = {k0..k0+7, k0+16..k0+23}
//                 lane>=16 -> M=lane-16, K shifted by +8
// Two aligned 16-B LDS loads per fragment.
// ---------------------------------------------------------------------------
__device__ __forceinline__ bf16x16 load_a_frag(const __bf16* __restrict__ base,
                                               int pitch, int lane, int k0) {
    const int m  = lane & 15;
    const int kb = k0 + ((lane >> 4) << 3);
    const __bf16* p = base + m * pitch + kb;
    bf16x8 lo = *(const bf16x8*)(p);
    bf16x8 hi = *(const bf16x8*)(p + 16);
    return __builtin_shufflevector(lo, hi, 0, 1, 2, 3, 4, 5, 6, 7,
                                   8, 9, 10, 11, 12, 13, 14, 15);
}

// B (32x16, KxN) read from TRANSPOSED stage wT[n][k] (K contiguous):
// lane -> N = n0 + (lane&15); K = {0..7, 16..23} shifted by +8 for lane>=16
__device__ __forceinline__ bf16x16 load_b_fragT(const __bf16* __restrict__ baseT,
                                                int lane, int n0) {
    const int n  = n0 + (lane & 15);
    const int kb = (lane >> 4) << 3;
    const __bf16* p = baseT + n * KPITCH + kb;
    bf16x8 lo = *(const bf16x8*)(p);
    bf16x8 hi = *(const bf16x8*)(p + 16);
    return __builtin_shufflevector(lo, hi, 0, 1, 2, 3, 4, 5, 6, 7,
                                   8, 9, 10, 11, 12, 13, 14, 15);
}

__device__ __forceinline__ float gelu_erf(float v) {
    return 0.5f * v * (1.0f + erff(v * 0.70710678118654752f));
}

// ---------------------------------------------------------------------------
// 4) Fused gathered MLP:  out[idx[m], :] += gelu(x[idx[m],:] @ Wfc + bfc) @ Wproj
//    grid.x : 16-token tiles (max 128), grid.y : d_ff range splits
//    block  : 256 threads = 8 wave32
// ---------------------------------------------------------------------------
__global__ __launch_bounds__(256)
void mlp_routed_kernel(const __bf16* __restrict__ xb,       // [N_TOK, 768] bf16
                       const int*   __restrict__ idx_list,  // compacted tokens
                       const int*   __restrict__ n_ptr,     // token count
                       const float* __restrict__ w_fc,      // [768, d_ff]
                       const float* __restrict__ b_fc,      // [d_ff]
                       const float* __restrict__ w_proj,    // [d_ff, 768]
                       float*       __restrict__ out,       // [N_TOK, 768]
                       int d_ff, int f_per_block) {
    __shared__ __align__(16) __bf16 xs[16 * XPITCH];        // ~24.3 KB x tile
    __shared__ __align__(16) __bf16 hs[16 * HPITCH];        // ~4.3 KB hidden
    __shared__ __align__(16) __bf16 wT[D_MODEL * KPITCH];   // 60 KB weight stage
    __shared__ int idx_s[16];

    const int tid  = threadIdx.x;
    const int wave = tid >> 5;
    const int lane = tid & 31;

    const int n_tok = *n_ptr;
    const int m0 = blockIdx.x * 16;
    if (m0 >= n_tok) return;
    const int rows = min(16, n_tok - m0);

    if (tid < 16) idx_s[tid] = idx_list[(tid < rows) ? (m0 + tid) : m0];
    __syncthreads();

    // gather 16 token rows of bf16 activations into LDS (coalesced per row)
    for (int e = tid; e < 16 * D_MODEL; e += 256) {
        int r = e / D_MODEL, c = e - r * D_MODEL;
        xs[r * XPITCH + c] = xb[(unsigned)idx_s[r] * D_MODEL + c];
    }

    const int f_base = blockIdx.y * f_per_block;
    const unsigned dffu = (unsigned)d_ff;

    floatx8 acc[6];
#pragma unroll
    for (int s = 0; s < 6; ++s) acc[s] = (floatx8){};

    for (int fc0 = 0; fc0 < f_per_block; fc0 += FCHUNK) {
        const int f0 = f_base + fc0;

        // ------------- FC phase: h = x @ Wfc[:, f0:f0+128] -------------------
        floatx8 hacc = (floatx8){};
        for (int k0 = 0; k0 < D_MODEL; k0 += 32) {
            __syncthreads();
            // stage Wfc[k0:k0+32, f0:f0+128] as TRANSPOSED bf16 wT[n][k]
            // 1024 float4 quads, 4 per thread, coalesced global_load_b128
#pragma unroll
            for (int it = 0; it < 4; ++it) {
                int q  = tid + it * 256;          // 0..1023
                int r  = q >> 5;                  // k row 0..31
                int n0 = (q & 31) << 2;           // n col 0..124
                float4 v = *(const float4*)&w_fc[(unsigned)(k0 + r) * dffu + f0 + n0];
                wT[(n0 + 0) * KPITCH + r] = (__bf16)v.x;
                wT[(n0 + 1) * KPITCH + r] = (__bf16)v.y;
                wT[(n0 + 2) * KPITCH + r] = (__bf16)v.z;
                wT[(n0 + 3) * KPITCH + r] = (__bf16)v.w;
            }
            if (k0 + 32 < D_MODEL)
                __builtin_prefetch(&w_fc[(unsigned)(k0 + 32) * dffu + f0 + (tid & 127)], 0, 1);
            __syncthreads();
            bf16x16 a = load_a_frag(xs, XPITCH, lane, k0);
            bf16x16 b = load_b_fragT(wT, lane, wave * 16);
            hacc = __builtin_amdgcn_wmma_f32_16x16x32_bf16(
                false, a, false, b, (short)0, hacc, false, false);
        }

        // bias + exact-erf GELU, write h tile (bf16) to LDS
        {
            const float bias = b_fc[f0 + wave * 16 + (lane & 15)];
            const int mbase = (lane >> 4) << 3;
#pragma unroll
            for (int r = 0; r < 8; ++r) {
                float v = gelu_erf(hacc[r] + bias);
                hs[(mbase + r) * HPITCH + wave * 16 + (lane & 15)] = (__bf16)v;
            }
        }

        // ------------- PROJ phase: acc += h @ Wproj[f0:f0+128, :] ------------
        for (int ks = 0; ks < 4; ++ks) {
            __syncthreads();
            // stage Wproj[f0+ks*32 : +32, 0:768] as TRANSPOSED bf16 wT[n][k]
            // 6144 float4 quads, 24 per thread
#pragma unroll
            for (int it = 0; it < 24; ++it) {
                int q  = tid + it * 256;          // 0..6143
                int r  = q / 192;                 // k row 0..31
                int n0 = (q - r * 192) << 2;      // n col 0..764
                float4 v = *(const float4*)&w_proj[(unsigned)(f0 + ks * 32 + r) * D_MODEL + n0];
                wT[(n0 + 0) * KPITCH + r] = (__bf16)v.x;
                wT[(n0 + 1) * KPITCH + r] = (__bf16)v.y;
                wT[(n0 + 2) * KPITCH + r] = (__bf16)v.z;
                wT[(n0 + 3) * KPITCH + r] = (__bf16)v.w;
            }
            if (ks < 3)
                __builtin_prefetch(&w_proj[(unsigned)(f0 + ks * 32 + 32) * D_MODEL + (tid << 2)], 0, 1);
            __syncthreads();
            bf16x16 a = load_a_frag(hs, HPITCH, lane, ks * 32);
#pragma unroll
            for (int s = 0; s < 6; ++s) {
                bf16x16 b = load_b_fragT(wT, lane, (wave * 6 + s) * 16);
                acc[s] = __builtin_amdgcn_wmma_f32_16x16x32_bf16(
                    false, a, false, b, (short)0, acc[s], false, false);
            }
        }
    }

    // ------------- write-out: atomic accumulate partial sums -----------------
    const int mbase = (lane >> 4) << 3;
#pragma unroll
    for (int s = 0; s < 6; ++s) {
        const int n = (wave * 6 + s) * 16 + (lane & 15);
#pragma unroll
        for (int r = 0; r < 8; ++r) {
            const int m = mbase + r;
            if (m < rows) {
                atomicAdd(&out[(unsigned)idx_s[m] * D_MODEL + n], acc[s][r]);
            }
        }
    }
}

// ---------------------------------------------------------------------------
// Launcher
// ---------------------------------------------------------------------------
extern "C" void kernel_launch(void* const* d_in, const int* in_sizes, int n_in,
                              void* d_out, int out_size, void* d_ws, size_t ws_size,
                              hipStream_t stream) {
    const float* x        = (const float*)d_in[0];
    const int*   mask     = (const int*)  d_in[1];
    const float* w_fc_s   = (const float*)d_in[2];
    const float* b_fc_s   = (const float*)d_in[3];
    const float* w_proj_s = (const float*)d_in[4];
    const float* b_proj_s = (const float*)d_in[5];
    const float* w_fc_b   = (const float*)d_in[6];
    const float* b_fc_b   = (const float*)d_in[7];
    const float* w_proj_b = (const float*)d_in[8];
    const float* b_proj_b = (const float*)d_in[9];
    float* out = (float*)d_out;

    // workspace layout
    char* ws = (char*)d_ws;
    int*     counts    = (int*)ws;                        // [0]=n_big [1]=n_small
    int*     big_idx   = (int*)(ws + 16);                 // 2048 ints
    int*     small_idx = (int*)(ws + 16 + 8192);          // 2048 ints
    __bf16*  xb        = (__bf16*)(ws + 16400);           // 2048*768 bf16

    route_kernel<<<1, 32, 0, stream>>>(mask, big_idx, small_idx, counts);

    const int n_x = N_TOK * D_MODEL;
    cast_x_kernel<<<(n_x + 255) / 256, 256, 0, stream>>>(x, xb, n_x);
    init_out_kernel<<<(n_x + 255) / 256, 256, 0, stream>>>(out, mask, b_proj_s, b_proj_b);

    // big expert: d_ff = 92160, 48 f-splits of 1920 (15 chunks of 128 each)
    mlp_routed_kernel<<<dim3(N_TOK / 16, 48), 256, 0, stream>>>(
        xb, big_idx, counts + 0, w_fc_b, b_fc_b, w_proj_b, out, 92160, 1920);

    // small expert: d_ff = 3072, 4 f-splits of 768 (6 chunks of 128 each)
    mlp_routed_kernel<<<dim3(N_TOK / 16, 4), 256, 0, stream>>>(
        xb, small_idx, counts + 1, w_fc_s, b_fc_s, w_proj_s, out, 3072, 768);
}
